// MultiHeadAttention_47201690583040
// MI455X (gfx1250) — matrix operand order repaired and checked by reading, not compiled, not run on hardware
//
#include <hip/hip_runtime.h>
#include <hip/hip_bf16.h>
#include <stdint.h>
#include <stddef.h>

// ---------------------------------------------------------------------------
// Types for gfx1250 WMMA (wave32, v_wmma_f32_16x16x32_bf16)
// ---------------------------------------------------------------------------
typedef __bf16 bf16_t;
typedef bf16_t v16bf __attribute__((ext_vector_type(16)));
typedef bf16_t v8bf  __attribute__((ext_vector_type(8)));
typedef float  v8f   __attribute__((ext_vector_type(8)));

#define NHEAD    16
#define HDIM     128
#define BSZ      2
#define TSEQ     2048
#define CDIM     2048

// ---------------------------------------------------------------------------
// Async global->LDS copy path (CDNA5 GLOBAL_LOAD_ASYNC_TO_LDS_B128, ASYNCcnt)
// Probe round 2 showed the builtin's signature: (int4 AS1*, int4 AS3*, Ii, Ii).
// AS1 pointer = 64-bit global address; AS3 pointer = low 32 bits of the
// generic address (aperture layout {base[63:32], lds_offset[31:0]}).
// ---------------------------------------------------------------------------
typedef int v4i __attribute__((vector_size(16)));
typedef __attribute__((address_space(1))) v4i as1_v4i;
typedef __attribute__((address_space(3))) v4i as3_v4i;

#if __has_builtin(__builtin_amdgcn_global_load_async_to_lds_b128)
#define HAVE_ASYNC_LDS 1
#define ASYNC_CP16(dst_lds, src_glb)                                          \
  __builtin_amdgcn_global_load_async_to_lds_b128(                             \
      (as1_v4i*)(unsigned long long)(uintptr_t)(src_glb),                     \
      (as3_v4i*)(unsigned int)(uintptr_t)(dst_lds), 0, 0)
#else
#define HAVE_ASYNC_LDS 0
#define ASYNC_CP16(dst_lds, src_glb)                                          \
  (*(uint4*)(dst_lds) = *(const uint4*)(src_glb))
#endif

#if __has_builtin(__builtin_amdgcn_s_wait_asynccnt)
#define ASYNC_WAIT_SOME() __builtin_amdgcn_s_wait_asynccnt(4)
#define ASYNC_WAIT_ALL()  __builtin_amdgcn_s_wait_asynccnt(0)
#elif HAVE_ASYNC_LDS
#define ASYNC_WAIT_SOME() asm volatile("s_wait_asynccnt 0x4" ::: "memory")
#define ASYNC_WAIT_ALL()  asm volatile("s_wait_asynccnt 0x0" ::: "memory")
#else
#define ASYNC_WAIT_SOME()
#define ASYNC_WAIT_ALL()
#endif

static __device__ __forceinline__ bf16_t f2bf(float f) {
  uint32_t u = __builtin_bit_cast(uint32_t, f);
  uint32_t r = u + 0x7FFFu + ((u >> 16) & 1u);   // round-to-nearest-even
  return __builtin_bit_cast(bf16_t, (uint16_t)(r >> 16));
}
static __device__ __forceinline__ float bf2f(bf16_t b) {
  uint32_t u = ((uint32_t)__builtin_bit_cast(uint16_t, b)) << 16;
  return __builtin_bit_cast(float, u);
}
static __device__ __forceinline__ uint16_t bfbits(float f) {
  uint32_t u = __builtin_bit_cast(uint32_t, f);
  uint32_t r = u + 0x7FFFu + ((u >> 16) & 1u);
  return (uint16_t)(r >> 16);
}

// ---------------------------------------------------------------------------
// Kernel 1: fp32 -> bf16 bulk convert (n multiple of 4)
// ---------------------------------------------------------------------------
__global__ __launch_bounds__(256) void cvt_f32_to_bf16(
    const float* __restrict__ in, bf16_t* __restrict__ out, int n) {
  int i = (blockIdx.x * 256 + threadIdx.x) * 4;
  if (i + 3 < n) {
    float4 v = *(const float4*)(in + i);
    out[i + 0] = f2bf(v.x);
    out[i + 1] = f2bf(v.y);
    out[i + 2] = f2bf(v.z);
    out[i + 3] = f2bf(v.w);
  }
}

// ---------------------------------------------------------------------------
// Kernel 2/5: TN GEMM  C[m,n] = sum_k A[m,k] * W[n,k]
//   A: M x K row-major bf16, W: N x K row-major bf16.
//   Block = 256 threads (8 waves), block tile 128(M) x 128(N),
//   wave grid 2x4, each wave 64x32 = 4x2 WMMA accumulators.
//   K-step 32, double-buffered LDS (rows padded to 36 elements = 72B so the
//   16 rows read by a fragment hit distinct banks: 18*row mod 64 distinct).
//   Tiles staged with GLOBAL_LOAD_ASYNC_TO_LDS_B128: 4 chunks/thread/buffer,
//   issue(next) -> s_wait_asynccnt<=4 -> barrier -> compute -> barrier.
// ---------------------------------------------------------------------------
#define LDSPITCH 36

template <bool OUT_F32>
__global__ __launch_bounds__(256) void gemm_tn_wmma(
    const bf16_t* __restrict__ A, const bf16_t* __restrict__ W,
    void* __restrict__ Cp, int M, int N, int K) {
  __shared__ bf16_t sA[2][128][LDSPITCH];
  __shared__ bf16_t sB[2][128][LDSPITCH];

  const int tid = threadIdx.x;
  const int lane = tid & 31;
  const int wave = tid >> 5;
  const int tilesN = N >> 7;
  const int m0 = (blockIdx.x / tilesN) << 7;
  const int n0 = (blockIdx.x % tilesN) << 7;
  const int g = lane >> 4;
  const int r16 = lane & 15;
  const int wm = (wave & 1) << 6;   // 0 / 64
  const int wn = (wave >> 1) << 5;  // 0 / 32 / 64 / 96

  // stage one 128x32 A tile + 128x32 B tile (16 KB) into buffer `buf`
  auto issue = [&](int buf, int k0) {
#pragma unroll
    for (int j = 0; j < 2; ++j) {
      int c = tid + 256 * j;          // 512 chunks of 16B per operand
      int row = c >> 2;
      int cc = (c & 3) << 3;          // element offset 0/8/16/24
      ASYNC_CP16(&sA[buf][row][cc], A + (size_t)(m0 + row) * K + k0 + cc);
      ASYNC_CP16(&sB[buf][row][cc], W + (size_t)(n0 + row) * K + k0 + cc);
    }
  };

  v8f acc[4][2];
#pragma unroll
  for (int mt = 0; mt < 4; ++mt)
#pragma unroll
    for (int nt = 0; nt < 2; ++nt) acc[mt][nt] = (v8f)0.0f;

  const int nk = K >> 5;
  issue(0, 0);
  for (int kb = 0; kb < nk; ++kb) {
    const int buf = kb & 1;
    if (kb + 1 < nk) {
      issue(buf ^ 1, (kb + 1) << 5);
      ASYNC_WAIT_SOME();              // oldest 4 (this buffer) are complete
    } else {
      ASYNC_WAIT_ALL();
    }
    __syncthreads();

    v16bf a[4];
#pragma unroll
    for (int mt = 0; mt < 4; ++mt) {
      v8bf lo = *(const v8bf*)&sA[buf][wm + mt * 16 + r16][16 * g];
      v8bf hi = *(const v8bf*)&sA[buf][wm + mt * 16 + r16][16 * g + 8];
#pragma unroll
      for (int i = 0; i < 8; ++i) { a[mt][i] = lo[i]; a[mt][8 + i] = hi[i]; }
    }
    v16bf b[2];
#pragma unroll
    for (int nt = 0; nt < 2; ++nt)
      b[nt] = *(const v16bf*)&sB[buf][wn + nt * 16 + r16][16 * g];
#pragma unroll
    for (int mt = 0; mt < 4; ++mt)
#pragma unroll
      for (int nt = 0; nt < 2; ++nt)
        acc[mt][nt] = __builtin_amdgcn_wmma_f32_16x16x32_bf16(
            false, a[mt], false, b[nt], (short)0, acc[mt][nt], false, false);

    __syncthreads();                  // buffer free for reuse
  }

  // C/D layout: VGPR r, lane -> row = r + 8*(lane/16), col = lane%16
#pragma unroll
  for (int mt = 0; mt < 4; ++mt)
#pragma unroll
    for (int nt = 0; nt < 2; ++nt)
#pragma unroll
      for (int r = 0; r < 8; ++r) {
        size_t row = (size_t)(m0 + wm + mt * 16 + r + 8 * g);
        size_t col = (size_t)(n0 + wn + nt * 16 + r16);
        if (OUT_F32)
          ((float*)Cp)[row * (size_t)N + col] = acc[mt][nt][r];
        else
          ((bf16_t*)Cp)[row * (size_t)N + col] = f2bf(acc[mt][nt][r]);
      }
}

// ---------------------------------------------------------------------------
// Kernel 3: RoPE + layout. Reads qkv bf16 [B*T, 3*C]; writes:
//   q,k : [B,H,T,D] bf16 (rotated),  vT : [B,H,D,T] bf16 (transposed copy)
// One thread per (b,t,h,i) with i in [0,64): handles pair (i, i+64).
// ---------------------------------------------------------------------------
__global__ __launch_bounds__(256) void rope_split(
    const bf16_t* __restrict__ qkv, bf16_t* __restrict__ q,
    bf16_t* __restrict__ k, bf16_t* __restrict__ vt) {
  int idx = blockIdx.x * 256 + threadIdx.x;   // 2^22 threads total
  int i = idx & 63;
  int h = (idx >> 6) & 15;
  int t = (idx >> 10) & 2047;
  int b = idx >> 21;

  // div[i] = exp(-i*ln(10000)/64) = 2^(-i*log2(10000)/64)
  float inv = exp2f(-(float)i * 0.2076205059304601f);
  float ang = (float)t * inv;
  float s = sinf(ang), c = cosf(ang);

  size_t row = (size_t)(b * TSEQ + t) * (size_t)(3 * CDIM);
  size_t hq = row + (size_t)h * HDIM;
  float q1 = bf2f(qkv[hq + i]);
  float q2 = bf2f(qkv[hq + 64 + i]);
  float k1 = bf2f(qkv[hq + CDIM + i]);
  float k2 = bf2f(qkv[hq + CDIM + 64 + i]);
  float v1 = bf2f(qkv[hq + 2 * CDIM + i]);
  float v2 = bf2f(qkv[hq + 2 * CDIM + 64 + i]);

  size_t bh = (size_t)(b * NHEAD + h);
  size_t qo = (bh * TSEQ + (size_t)t) * HDIM;
  q[qo + i]      = f2bf(q1 * c - q2 * s);
  q[qo + 64 + i] = f2bf(q2 * c + q1 * s);
  k[qo + i]      = f2bf(k1 * c - k2 * s);
  k[qo + 64 + i] = f2bf(k2 * c + k1 * s);

  size_t vo = bh * (size_t)HDIM * TSEQ + (size_t)t;
  vt[vo + (size_t)i * TSEQ]        = f2bf(v1);
  vt[vo + (size_t)(i + 64) * TSEQ] = f2bf(v2);
}

// ---------------------------------------------------------------------------
// Kernel 4: causal flash attention.
// One wave per (b,h,16-query-row block). Q fragments resident (16x128).
// Per 32-key step: 8 WMMA (scores, K=128) + online softmax + 8 WMMA (PV).
// P is re-laid-out D-format -> A-format through a padded LDS tile.
// Output written bf16 into attn [B*T, C].
// ---------------------------------------------------------------------------
__global__ __launch_bounds__(256) void flash_attn_wmma(
    const bf16_t* __restrict__ Q, const bf16_t* __restrict__ Kt,
    const bf16_t* __restrict__ Vt, bf16_t* __restrict__ Oa) {
  __shared__ unsigned short pst[8][16][40];   // per-wave 16x32 (+8 pad) bf16

  const int lane = threadIdx.x & 31;
  const int wave = threadIdx.x >> 5;
  const int tile = blockIdx.x * 8 + wave;     // 4096 tiles
  const int tblk = tile & 127;
  const int h = (tile >> 7) & 15;
  const int b = tile >> 11;
  const int t0 = tblk << 4;
  const int g = lane >> 4;
  const int r16 = lane & 15;

  const size_t bh = (size_t)(b * NHEAD + h);
  const bf16_t* q = Q + bh * (size_t)TSEQ * HDIM;
  const bf16_t* kk = Kt + bh * (size_t)TSEQ * HDIM;
  const bf16_t* vt = Vt + bh * (size_t)HDIM * TSEQ;

  // resident Q A-fragments (4 x K=32 covering D=128)
  v16bf qf[4];
  {
    const bf16_t* qrow = q + (size_t)(t0 + r16) * HDIM;
#pragma unroll
    for (int c = 0; c < 4; ++c) {
      v8bf lo = *(const v8bf*)(qrow + c * 32 + 16 * g);
      v8bf hi = *(const v8bf*)(qrow + c * 32 + 16 * g + 8);
#pragma unroll
      for (int i = 0; i < 8; ++i) { qf[c][i] = lo[i]; qf[c][8 + i] = hi[i]; }
    }
  }

  float rm[8], rl[8];
  v8f o[8];
#pragma unroll
  for (int r = 0; r < 8; ++r) { rm[r] = -3.0e38f; rl[r] = 0.0f; }
#pragma unroll
  for (int n = 0; n < 8; ++n) o[n] = (v8f)0.0f;

  const float sc = 0.08838834764831845f * 1.4426950408889634f; // 1/sqrt(128)*log2e

  for (int s0 = 0; s0 < t0 + 16; s0 += 32) {
    v8f sa = (v8f)0.0f, sb = (v8f)0.0f;
#pragma unroll
    for (int c = 0; c < 4; ++c) {
      v16bf kf0 = *(const v16bf*)(kk + (size_t)(s0 + r16) * HDIM + c * 32 + 16 * g);
      v16bf kf1 = *(const v16bf*)(kk + (size_t)(s0 + 16 + r16) * HDIM + c * 32 + 16 * g);
      sa = __builtin_amdgcn_wmma_f32_16x16x32_bf16(false, qf[c], false, kf0,
                                                   (short)0, sa, false, false);
      sb = __builtin_amdgcn_wmma_f32_16x16x32_bf16(false, qf[c], false, kf1,
                                                   (short)0, sb, false, false);
    }

    float tmax[8];
#pragma unroll
    for (int r = 0; r < 8; ++r) {
      int row = t0 + r + 8 * g;
      float x0 = (s0 + r16 <= row) ? sa[r] * sc : -3.0e38f;
      float x1 = (s0 + 16 + r16 <= row) ? sb[r] * sc : -3.0e38f;
      sa[r] = x0; sb[r] = x1;
      tmax[r] = fmaxf(x0, x1);
    }
#pragma unroll
    for (int off = 1; off < 16; off <<= 1)
#pragma unroll
      for (int r = 0; r < 8; ++r)
        tmax[r] = fmaxf(tmax[r], __shfl_xor(tmax[r], off, 32));

    float al[8], tsum[8];
#pragma unroll
    for (int r = 0; r < 8; ++r) {
      float nm = fmaxf(rm[r], tmax[r]);
      al[r] = exp2f(rm[r] - nm);
      rm[r] = nm;
      float p0 = exp2f(sa[r] - nm);
      float p1 = exp2f(sb[r] - nm);
      sa[r] = p0; sb[r] = p1;
      tsum[r] = p0 + p1;
    }
#pragma unroll
    for (int off = 1; off < 16; off <<= 1)
#pragma unroll
      for (int r = 0; r < 8; ++r)
        tsum[r] += __shfl_xor(tsum[r], off, 32);
#pragma unroll
    for (int r = 0; r < 8; ++r) rl[r] = rl[r] * al[r] + tsum[r];
#pragma unroll
    for (int n = 0; n < 8; ++n)
#pragma unroll
      for (int r = 0; r < 8; ++r) o[n][r] *= al[r];

    // D-layout -> A-layout via LDS
    unsigned short (*ps)[40] = pst[wave];
#pragma unroll
    for (int r = 0; r < 8; ++r) {
      int row = r + 8 * g;
      ps[row][r16] = bfbits(sa[r]);
      ps[row][16 + r16] = bfbits(sb[r]);
    }
    asm volatile("s_wait_dscnt 0" ::: "memory");
    v16bf pf;
#pragma unroll
    for (int j = 0; j < 8; ++j) {
      pf[j] = __builtin_bit_cast(bf16_t, ps[r16][16 * g + j]);
      pf[8 + j] = __builtin_bit_cast(bf16_t, ps[r16][16 * g + 8 + j]);
    }

    // O += P @ V   (V B-frags contiguous from transposed vT)
#pragma unroll
    for (int n = 0; n < 8; ++n) {
      v16bf vf = *(const v16bf*)(vt + (size_t)(n * 16 + r16) * TSEQ + s0 + 16 * g);
      o[n] = __builtin_amdgcn_wmma_f32_16x16x32_bf16(false, pf, false, vf,
                                                     (short)0, o[n], false, false);
    }
  }

#pragma unroll
  for (int n = 0; n < 8; ++n)
#pragma unroll
    for (int r = 0; r < 8; ++r) {
      float val = o[n][r] / rl[r];
      size_t row = (size_t)(b * TSEQ + t0 + r + 8 * g);
      size_t col = (size_t)(h * HDIM + n * 16 + r16);
      Oa[row * CDIM + col] = f2bf(val);
    }
}

// ---------------------------------------------------------------------------
// Host launcher
// ---------------------------------------------------------------------------
extern "C" void kernel_launch(void* const* d_in, const int* in_sizes, int n_in,
                              void* d_out, int out_size, void* d_ws, size_t ws_size,
                              hipStream_t stream) {
  (void)in_sizes; (void)n_in; (void)out_size; (void)ws_size;
  const float* x = (const float*)d_in[0];       // [B,T,C]
  const float* qkv_w = (const float*)d_in[1];   // [3C,C]
  const float* out_w = (const float*)d_in[2];   // [C,C]
  float* out = (float*)d_out;                   // [B,T,C]

  const int M = BSZ * TSEQ;                     // 4096
  char* ws = (char*)d_ws;
  size_t off = 0;
  auto take = [&](size_t bytes) {
    void* p = ws + off;
    off += (bytes + 255) & ~(size_t)255;
    return p;
  };
  bf16_t* xb    = (bf16_t*)take((size_t)M * CDIM * 2);            // 16 MB
  bf16_t* wb    = (bf16_t*)take((size_t)3 * CDIM * CDIM * 2);     // 24 MB
  bf16_t* ob    = (bf16_t*)take((size_t)CDIM * CDIM * 2);         //  8 MB
  bf16_t* qkvb  = (bf16_t*)take((size_t)M * 3 * CDIM * 2);        // 48 MB
  bf16_t* qb    = (bf16_t*)take((size_t)M * CDIM * 2);            // 16 MB
  bf16_t* kb    = (bf16_t*)take((size_t)M * CDIM * 2);            // 16 MB
  bf16_t* vtb   = (bf16_t*)take((size_t)M * CDIM * 2);            // 16 MB
  bf16_t* attnb = (bf16_t*)take((size_t)M * CDIM * 2);            // 16 MB

  // 1) fp32 -> bf16
  cvt_f32_to_bf16<<<(M * CDIM) / 1024, 256, 0, stream>>>(x, xb, M * CDIM);
  cvt_f32_to_bf16<<<(3 * CDIM * CDIM) / 1024, 256, 0, stream>>>(qkv_w, wb, 3 * CDIM * CDIM);
  cvt_f32_to_bf16<<<(CDIM * CDIM) / 1024, 256, 0, stream>>>(out_w, ob, CDIM * CDIM);

  // 2) QKV projection: [4096,2048] x [6144,2048]^T -> bf16 [4096,6144]
  {
    int blocks = (M / 128) * ((3 * CDIM) / 128);   // 32*48 = 1536
    gemm_tn_wmma<false><<<blocks, 256, 0, stream>>>(xb, wb, qkvb, M, 3 * CDIM, CDIM);
  }

  // 3) RoPE + head split + V transpose
  rope_split<<<(BSZ * TSEQ * NHEAD * 64) / 256, 256, 0, stream>>>(qkvb, qb, kb, vtb);

  // 4) causal flash attention -> bf16 attn [4096,2048]
  {
    int tiles = BSZ * NHEAD * (TSEQ / 16);      // 4096 waves
    flash_attn_wmma<<<tiles / 8, 256, 0, stream>>>(qb, kb, vtb, attnb);
  }

  // 5) output projection: [4096,2048] x [2048,2048]^T -> f32 d_out
  {
    int blocks = (M / 128) * (CDIM / 128);      // 32*16 = 512
    gemm_tn_wmma<true><<<blocks, 256, 0, stream>>>(attnb, ob, out, M, CDIM, CDIM);
  }
}
